// UNetV2_concat_7430293422602
// MI455X (gfx1250) — matrix operand order repaired and checked by reading, not compiled, not run on hardware
//
#include <hip/hip_runtime.h>
#include <hip/hip_bf16.h>
#include <math.h>

// ---------------------------------------------------------------------------
// CDNA5 / gfx1250 sparse-UNet implementation.
// Core compute: gather-GEMM via v_wmma_f32_16x16x32_bf16 (wave32, one wave
// per 16x16 output tile, f32 accumulation, bf16 operands).
// Cin/Cout are template parameters -> branchless b128 gathers, shift-based
// tile math, immediate-offset stores with a wave-uniform full-tile fast path.
// ---------------------------------------------------------------------------

typedef __attribute__((ext_vector_type(16))) __bf16 v16bf;
typedef __attribute__((ext_vector_type(8)))  float  v8f;

#define KTAPS 27

static inline int cdiv(int a, int b) { return (a + b - 1) / b; }

// ---------------------------------------------------------------------------
// Pack fp32 weights w[27][Cin][Cout] -> bf16 tiles in WMMA B-operand layout.
// B is 32x16 (KxN) per tile. Per ISA 7.12.4/7.12.2 B striping:
//   lane n   (0..15): column n,  VGPR v half h -> K = 2v+h (v<4), 8+2v+h (v>=4)  (K = 0..7, 16..23)
//   lane n+16       : column n,  VGPR v half h -> K = 8+2v+h (v<4), 16+2v+h (v>=4) (K = 8..15, 24..31)
// Tile index = (k*KC + kc)*NC + nc ; each tile = 32 lanes * 16 halves.
// ---------------------------------------------------------------------------
__global__ void pack_w_bf16(const float* __restrict__ w, __bf16* __restrict__ wbuf,
                            int Cin, int Cout, int KC, int NC) {
  int total = KTAPS * KC * NC * 512;
  int tid = blockIdx.x * blockDim.x + threadIdx.x;
  if (tid >= total) return;
  int r    = tid & 15;          // half index within lane (== 2v+h)
  int lane = (tid >> 4) & 31;
  int tile = tid >> 9;
  int nc = tile % NC;
  int kc = (tile / NC) % KC;
  int k  = tile / (NC * KC);
  int col = nc * 16 + (lane & 15);
  int K   = ((lane & 16) ? 8 : 0) + (r < 8 ? r : r + 8);
  int cin = kc * 32 + K;
  float v = 0.0f;
  if (cin < Cin) v = w[((size_t)k * Cin + cin) * Cout + col];
  wbuf[(size_t)tile * 512 + lane * 16 + r] = (__bf16)v;
}

// ---------------------------------------------------------------------------
// Gather-GEMM: out[Nout x Cout] = sum_k gather(f, nb[:,k]) @ w[k]
// One wave per (16-row, 16-col) output tile. A-fragment layout (ISA 7.12.2,
// 16-bit A 16x32): lane l<16 holds row l, K in {0..7, 16..23}; lane l+16
// holds row l, K in {8..15, 24..31}. C/D layout: VGPR r, lanes 0-15 -> M=r,
// N=lane; lanes 16-31 -> M=8+r, N=lane-16.
// ---------------------------------------------------------------------------
__device__ __forceinline__ float4 ld4(const float* p) { return *(const float4*)p; }
__device__ __forceinline__ float4 mul4(float4 a, float s) {
  float4 r; r.x = a.x * s; r.y = a.y * s; r.z = a.z * s; r.w = a.w * s; return r;
}

template <int CIN, int COUT>
__global__ void __launch_bounds__(256)
conv_gather_wmma(const float* __restrict__ f, const int* __restrict__ nb,
                 const __bf16* __restrict__ wbuf, float* __restrict__ out,
                 int Nout, int nTiles) {
  constexpr int KC = (CIN + 31) / 32;
  constexpr int NC = COUT / 16;
  int wave = blockIdx.x * (blockDim.x >> 5) + (threadIdx.x >> 5);
  if (wave >= nTiles) return;              // wave-uniform: EXEC stays all-1s
  int lane = threadIdx.x & 31;
  int rowTile = wave / NC;                 // NC constexpr -> shift
  int nc      = wave % NC;
  int m   = lane & 15;
  int row = rowTile * 16 + m;
  int rowC = (row < Nout) ? row : (Nout - 1);    // clamped: loads always legal
  int hiK = (lane & 16) ? 8 : 0;           // A K-offset select per half-wave
  const __bf16* wlane = wbuf + lane * 16;

  v8f acc = {};
  for (int k = 0; k < KTAPS; ++k) {
    int idx = nb[(size_t)rowC * KTAPS + k];            // unconditional load
    float scale = (idx >= 0 && row < Nout) ? 1.0f : 0.0f;
    const float* src = f + (size_t)(idx > 0 ? idx : 0) * CIN;

#pragma unroll
    for (int kc = 0; kc < KC; ++kc) {
      // halves 0..7  -> K = kc*32 + hiK + j ; halves 8..15 -> +16
      float4 r0a, r0b, r1a, r1b;
      if constexpr (CIN == 4) {
        float4 t = ld4(src);                           // channels 0..3
        float msk = (hiK == 0) ? scale : 0.0f;         // lanes>=16: K=8.. invalid
        r0a = mul4(t, msk);
        r0b = make_float4(0.f, 0.f, 0.f, 0.f);
        r1a = r0b; r1b = r0b;
      } else if constexpr (CIN == 16) {
        // run0 (K = hiK..hiK+7) valid; run1 (K = 16+hiK..) out of range
        r0a = mul4(ld4(src + hiK), scale);
        r0b = mul4(ld4(src + hiK + 4), scale);
        r1a = make_float4(0.f, 0.f, 0.f, 0.f);
        r1b = r1a;
      } else {
        const float* s0 = src + kc * 32 + hiK;
        r0a = mul4(ld4(s0), scale);
        r0b = mul4(ld4(s0 + 4), scale);
        r1a = mul4(ld4(s0 + 16), scale);
        r1b = mul4(ld4(s0 + 20), scale);
      }

      v16bf a;
      a[0]  = (__bf16)r0a.x; a[1]  = (__bf16)r0a.y;
      a[2]  = (__bf16)r0a.z; a[3]  = (__bf16)r0a.w;
      a[4]  = (__bf16)r0b.x; a[5]  = (__bf16)r0b.y;
      a[6]  = (__bf16)r0b.z; a[7]  = (__bf16)r0b.w;
      a[8]  = (__bf16)r1a.x; a[9]  = (__bf16)r1a.y;
      a[10] = (__bf16)r1a.z; a[11] = (__bf16)r1a.w;
      a[12] = (__bf16)r1b.x; a[13] = (__bf16)r1b.y;
      a[14] = (__bf16)r1b.z; a[15] = (__bf16)r1b.w;

      v16bf b = *(const v16bf*)(wlane + (size_t)((k * KC + kc) * NC + nc) * 512);
      acc = __builtin_amdgcn_wmma_f32_16x16x32_bf16(
          false, a, false, b, (short)0, acc, false, false);
    }
  }

  int orowBase = rowTile * 16 + ((lane & 16) ? 8 : 0);
  int col = nc * 16 + m;
  float* op = out + (size_t)orowBase * COUT + col;     // immediate r*COUT offsets
  if (rowTile * 16 + 16 <= Nout) {
    // full tile: wave-uniform fast path, no per-row guards
#pragma unroll
    for (int r = 0; r < 8; ++r) op[r * COUT] = acc[r];
  } else {
#pragma unroll
    for (int r = 0; r < 8; ++r) {
      if (orowBase + r < Nout) op[r * COUT] = acc[r];
    }
  }
}

// ---------------------------------------------------------------------------
// vf = voxels.sum(1) / max(vnp,1) ; voxels: (N,5,4)
// ---------------------------------------------------------------------------
__global__ void voxel_mean_k(const float* __restrict__ vox,
                             const int* __restrict__ vnp,
                             float* __restrict__ vf, int N) {
  int tid = blockIdx.x * blockDim.x + threadIdx.x;
  if (tid >= N * 4) return;
  int n = tid >> 2, c = tid & 3;
  float s = 0.0f;
#pragma unroll
  for (int t = 0; t < 5; ++t) s += vox[((size_t)n * 5 + t) * 4 + c];
  int d = vnp[n]; if (d < 1) d = 1;
  vf[tid] = s / (float)d;
}

// ---------------------------------------------------------------------------
// BN stats per column: stats[c]=mean, stats[C+c]=rsqrt(var+1e-3)
// ---------------------------------------------------------------------------
__global__ void bn_stats_k(const float* __restrict__ x, float* __restrict__ stats,
                           int N, int C) {
  int c = blockIdx.x;
  float s = 0.0f, s2 = 0.0f;
  for (int n = threadIdx.x; n < N; n += blockDim.x) {
    float v = x[(size_t)n * C + c];
    s += v; s2 += v * v;
  }
  __shared__ float sh[256], sh2[256];
  sh[threadIdx.x] = s; sh2[threadIdx.x] = s2;
  __syncthreads();
  for (int off = 128; off > 0; off >>= 1) {
    if ((int)threadIdx.x < off) {
      sh[threadIdx.x]  += sh[threadIdx.x + off];
      sh2[threadIdx.x] += sh2[threadIdx.x + off];
    }
    __syncthreads();
  }
  if (threadIdx.x == 0) {
    float inv = 1.0f / (float)N;
    float mu  = sh[0] * inv;
    float var = sh2[0] * inv - mu * mu;
    stats[c]     = mu;
    stats[C + c] = rsqrtf(var + 1e-3f);
  }
}

// y = (x-mu)*istd*g + b [+ res] [relu]
__global__ void bn_apply_k(const float* __restrict__ x, const float* __restrict__ res,
                           const float* __restrict__ stats,
                           const float* __restrict__ g, const float* __restrict__ b,
                           float* __restrict__ y, int N, int C, int doRelu) {
  int tid = blockIdx.x * blockDim.x + threadIdx.x;
  if (tid >= N * C) return;
  int c = tid % C;
  float v = (x[tid] - stats[c]) * stats[C + c] * g[c] + b[c];
  if (res) v += res[tid];
  if (doRelu) v = fmaxf(v, 0.0f);
  y[tid] = v;
}

// cat = [a | b] along columns
__global__ void concat2_k(const float* __restrict__ a, const float* __restrict__ bsrc,
                          float* __restrict__ cat, int N, int Ca, int Cb) {
  int C = Ca + Cb;
  int tid = blockIdx.x * blockDim.x + threadIdx.x;
  if (tid >= N * C) return;
  int n = tid / C, c = tid % C;
  cat[tid] = (c < Ca) ? a[(size_t)n * Ca + c] : bsrc[(size_t)n * Cb + (c - Ca)];
}

// z[n,c] = xm[n,c] + cat[n,2c] + cat[n,2c+1]   (cat width = 2C)
__global__ void add_pairsum_k(const float* __restrict__ xm, const float* __restrict__ cat,
                              float* __restrict__ z, int N, int C) {
  int tid = blockIdx.x * blockDim.x + threadIdx.x;
  if (tid >= N * C) return;
  int n = tid / C, c = tid % C;
  const float* cr = cat + (size_t)n * 2 * C;
  z[tid] = xm[tid] + cr[2 * c] + cr[2 * c + 1];
}

// segment-max over batch id: out[b*outStride + colBase + c]
__global__ void seg_max_k(const float* __restrict__ f, const int* __restrict__ bid,
                          float* __restrict__ out, int N, int C,
                          int outStride, int colBase) {
  int b = blockIdx.x / C, c = blockIdx.x % C;
  float m = -INFINITY;
  for (int n = threadIdx.x; n < N; n += blockDim.x)
    if (bid[n] == b) m = fmaxf(m, f[(size_t)n * C + c]);
  __shared__ float sh[256];
  sh[threadIdx.x] = m;
  __syncthreads();
  for (int off = 128; off > 0; off >>= 1) {
    if ((int)threadIdx.x < off) sh[threadIdx.x] = fmaxf(sh[threadIdx.x], sh[threadIdx.x + off]);
    __syncthreads();
  }
  if (threadIdx.x == 0) out[(size_t)b * outStride + colBase + c] = sh[0];
}

// ---------------------------------------------------------------------------
// Host orchestration
// ---------------------------------------------------------------------------
namespace {

struct ConvP { const float* w; const float* g; const float* b; int Cin; int Cout; };

template <int CIN, int COUT>
void run_conv(hipStream_t s, const float* f, const int* nb, const __bf16* wbuf,
              float* outbuf, int Nout) {
  int nTiles = cdiv(Nout, 16) * (COUT / 16);
  conv_gather_wmma<CIN, COUT><<<cdiv(nTiles, 8), 256, 0, s>>>(
      f, nb, wbuf, outbuf, Nout, nTiles);
}

void launch_conv(hipStream_t s, __bf16* wbuf, const float* f, const int* nb,
                 int Nout, const ConvP& p, float* outbuf) {
  int KC = cdiv(p.Cin, 32), NC = p.Cout / 16;
  int packN = KTAPS * KC * NC * 512;
  pack_w_bf16<<<cdiv(packN, 256), 256, 0, s>>>(p.w, wbuf, p.Cin, p.Cout, KC, NC);
  switch (p.Cin * 1000 + p.Cout) {
    case   4016: run_conv<  4, 16>(s, f, nb, wbuf, outbuf, Nout); break;
    case  16016: run_conv< 16, 16>(s, f, nb, wbuf, outbuf, Nout); break;
    case  16032: run_conv< 16, 32>(s, f, nb, wbuf, outbuf, Nout); break;
    case  32016: run_conv< 32, 16>(s, f, nb, wbuf, outbuf, Nout); break;
    case  32032: run_conv< 32, 32>(s, f, nb, wbuf, outbuf, Nout); break;
    case  32064: run_conv< 32, 64>(s, f, nb, wbuf, outbuf, Nout); break;
    case  64032: run_conv< 64, 32>(s, f, nb, wbuf, outbuf, Nout); break;
    case  64064: run_conv< 64, 64>(s, f, nb, wbuf, outbuf, Nout); break;
    default:     run_conv<128, 64>(s, f, nb, wbuf, outbuf, Nout); break;
  }
}

void launch_block(hipStream_t s, __bf16* wbuf, float* stats, float* scr,
                  const float* f, const int* nb, int Nout,
                  const ConvP& p, const float* res, int relu, float* dest) {
  launch_conv(s, wbuf, f, nb, Nout, p, scr);
  bn_stats_k<<<p.Cout, 256, 0, s>>>(scr, stats, Nout, p.Cout);
  int tot = Nout * p.Cout;
  bn_apply_k<<<cdiv(tot, 256), 256, 0, s>>>(scr, res, stats, p.g, p.b, dest,
                                            Nout, p.Cout, relu);
}

// _ur(lat, bot, nb, inb, pt, pm, pinv)
void launch_ur(hipStream_t s, __bf16* wbuf, float* stats, float* scr,
               float* tb, float* catb, float* xmb,
               const float* lat, const float* bot, const int* nb, const int* inb,
               int N, int Nup, int C,
               const ConvP& t1c, const ConvP& t2c, const ConvP& mC, const ConvP& invC,
               float* dest) {
  // xt = bblock(lat)
  launch_block(s, wbuf, stats, scr, lat, nb, N, t1c, nullptr, 1, tb);
  launch_block(s, wbuf, stats, scr, tb,  nb, N, t2c, lat,     1, tb);
  // cat = [bot, xt]
  int tot = N * 2 * C;
  concat2_k<<<cdiv(tot, 256), 256, 0, s>>>(bot, tb, catb, N, C, C);
  // xm = block(cat)
  launch_block(s, wbuf, stats, scr, catb, nb, N, mC, nullptr, 1, xmb);
  // z = xm + pairsum(cat)
  add_pairsum_k<<<cdiv(N * C, 256), 256, 0, s>>>(xmb, catb, tb, N, C);
  // out = block(z, inb)
  launch_block(s, wbuf, stats, scr, tb, inb, Nup, invC, nullptr, 1, dest);
}

} // namespace

extern "C" void kernel_launch(void* const* d_in, const int* in_sizes, int n_in,
                              void* d_out, int out_size, void* d_ws, size_t ws_size,
                              hipStream_t stream) {
  // Input indices: pytree flatten with sorted dict keys.
  enum {
    IN_VOXELS = 0, IN_VNP = 1,
    P_C1 = 2, P_C2A = 5, P_C2B = 8, P_C2C = 11,
    P_C3A = 14, P_C3B = 17, P_C3C = 20,
    P_C4A = 23, P_C4B = 26, P_C4C = 29,
    P_C5 = 32, P_CIN = 35,
    P_I2 = 38, P_I3 = 41, P_I4 = 44,
    P_M1 = 47, P_M2 = 50, P_M3 = 53, P_M4 = 56,
    P_T1 = 59, P_T2 = 65, P_T3 = 71, P_T4 = 77,
    IN_NB1 = 83, IN_NB2D = 84, IN_NB2 = 85, IN_NB3D = 86, IN_NB3 = 87,
    IN_NB4D = 88, IN_NB4 = 89, IN_INB4 = 90, IN_INB3 = 91, IN_INB2 = 92,
    IN_BID1 = 93, IN_BID2 = 94, IN_BID3 = 95
  };
  (void)n_in; (void)ws_size;

  auto CF = [&](int i) { return (const float*)d_in[i]; };
  auto CI = [&](int i) { return (const int*)d_in[i]; };
  // conv_p leaves sorted: b, g, w
  auto CP = [&](int base, int cin, int cout) {
    ConvP p; p.b = CF(base); p.g = CF(base + 1); p.w = CF(base + 2);
    p.Cin = cin; p.Cout = cout; return p;
  };
  // blk_p leaves sorted: b1, b2, g1, g2, w1, w2
  auto BP1 = [&](int base, int c) {
    ConvP p; p.b = CF(base); p.g = CF(base + 2); p.w = CF(base + 4);
    p.Cin = c; p.Cout = c; return p;
  };
  auto BP2 = [&](int base, int c) {
    ConvP p; p.b = CF(base + 1); p.g = CF(base + 3); p.w = CF(base + 5);
    p.Cin = c; p.Cout = c; return p;
  };

  const int N1 = in_sizes[IN_BID1];
  const int N2 = in_sizes[IN_BID2];
  const int N3 = in_sizes[IN_BID3];
  const int N4 = in_sizes[IN_NB4] / KTAPS;
  const int B  = out_size / 128;

  // bump allocator over d_ws
  size_t off = 0;
  auto allocB = [&](size_t bytes) {
    void* p = (char*)d_ws + off;
    off += (bytes + 255) & ~(size_t)255;
    return p;
  };
  auto allocF = [&](size_t n) { return (float*)allocB(n * sizeof(float)); };
  auto maxi = [](size_t a, size_t b) { return a > b ? a : b; };

  size_t scrE = maxi(maxi((size_t)N1 * 16, (size_t)N2 * 32),
                     maxi((size_t)N3 * 64, (size_t)N4 * 64));
  size_t catE = maxi(maxi((size_t)N1 * 32, (size_t)N2 * 64),
                     maxi((size_t)N3 * 128, (size_t)N4 * 128));

  float* vf  = allocF((size_t)N1 * 4);
  float* xb  = allocF((size_t)N1 * 16);
  float* x1b = allocF((size_t)N1 * 16);
  float* x2b = allocF((size_t)N2 * 32);
  float* x3b = allocF((size_t)N3 * 64);
  float* x4b = allocF((size_t)N4 * 64);
  float* u4b = allocF((size_t)N3 * 64);
  float* u3b = allocF((size_t)N2 * 32);
  float* u2b = allocF((size_t)N1 * 16);
  float* u1b = allocF((size_t)N1 * 16);
  float* scr  = allocF(scrE);
  float* tb   = allocF(scrE);
  float* xmb  = allocF(scrE);
  float* catb = allocF(catE);
  float* stats = allocF(256);
  __bf16* wbuf = (__bf16*)allocB((size_t)KTAPS * 4 * 4 * 512 * sizeof(__bf16));

  const int* nb1  = CI(IN_NB1);
  const int* nb2d = CI(IN_NB2D); const int* nb2 = CI(IN_NB2);
  const int* nb3d = CI(IN_NB3D); const int* nb3 = CI(IN_NB3);
  const int* nb4d = CI(IN_NB4D); const int* nb4 = CI(IN_NB4);
  const int* inb4 = CI(IN_INB4); const int* inb3 = CI(IN_INB3);
  const int* inb2 = CI(IN_INB2);

  // ---- encoder ----
  voxel_mean_k<<<cdiv(N1 * 4, 256), 256, 0, stream>>>(CF(IN_VOXELS), CI(IN_VNP), vf, N1);
  launch_block(stream, wbuf, stats, scr, vf,  nb1,  N1, CP(P_CIN, 4, 16),  nullptr, 1, xb);
  launch_block(stream, wbuf, stats, scr, xb,  nb1,  N1, CP(P_C1, 16, 16),  nullptr, 1, x1b);
  launch_block(stream, wbuf, stats, scr, x1b, nb2d, N2, CP(P_C2A, 16, 32), nullptr, 1, x2b);
  launch_block(stream, wbuf, stats, scr, x2b, nb2,  N2, CP(P_C2B, 32, 32), nullptr, 1, x2b);
  launch_block(stream, wbuf, stats, scr, x2b, nb2,  N2, CP(P_C2C, 32, 32), nullptr, 1, x2b);
  launch_block(stream, wbuf, stats, scr, x2b, nb3d, N3, CP(P_C3A, 32, 64), nullptr, 1, x3b);
  launch_block(stream, wbuf, stats, scr, x3b, nb3,  N3, CP(P_C3B, 64, 64), nullptr, 1, x3b);
  launch_block(stream, wbuf, stats, scr, x3b, nb3,  N3, CP(P_C3C, 64, 64), nullptr, 1, x3b);
  launch_block(stream, wbuf, stats, scr, x3b, nb4d, N4, CP(P_C4A, 64, 64), nullptr, 1, x4b);
  launch_block(stream, wbuf, stats, scr, x4b, nb4,  N4, CP(P_C4B, 64, 64), nullptr, 1, x4b);
  launch_block(stream, wbuf, stats, scr, x4b, nb4,  N4, CP(P_C4C, 64, 64), nullptr, 1, x4b);

  // ---- decoder (_ur blocks) ----
  launch_ur(stream, wbuf, stats, scr, tb, catb, xmb, x4b, x4b, nb4, inb4,
            N4, N3, 64, BP1(P_T4, 64), BP2(P_T4, 64),
            CP(P_M4, 128, 64), CP(P_I4, 64, 64), u4b);
  launch_ur(stream, wbuf, stats, scr, tb, catb, xmb, x3b, u4b, nb3, inb3,
            N3, N2, 64, BP1(P_T3, 64), BP2(P_T3, 64),
            CP(P_M3, 128, 64), CP(P_I3, 64, 32), u3b);
  launch_ur(stream, wbuf, stats, scr, tb, catb, xmb, x2b, u3b, nb2, inb2,
            N2, N1, 32, BP1(P_T2, 32), BP2(P_T2, 32),
            CP(P_M2, 64, 32), CP(P_I2, 32, 16), u2b);
  launch_ur(stream, wbuf, stats, scr, tb, catb, xmb, x1b, u2b, nb1, nb1,
            N1, N1, 16, BP1(P_T1, 16), BP2(P_T1, 16),
            CP(P_M1, 32, 16), CP(P_C5, 16, 16), u1b);

  // ---- segment-max pooling -> (B, 128) ----
  float* outp = (float*)d_out;
  seg_max_k<<<B * 64, 256, 0, stream>>>(u4b, CI(IN_BID3), outp, N3, 64, 128, 0);
  seg_max_k<<<B * 32, 256, 0, stream>>>(u3b, CI(IN_BID2), outp, N2, 32, 128, 64);
  seg_max_k<<<B * 16, 256, 0, stream>>>(u2b, CI(IN_BID1), outp, N1, 16, 128, 96);
  seg_max_k<<<B * 16, 256, 0, stream>>>(u1b, CI(IN_BID1), outp, N1, 16, 128, 112);
}
